// HeteroEdgePredictor_per_node_13769665151131
// MI455X (gfx1250) — matrix operand order repaired and checked by reading, not compiled, not run on hardware
//
#include <hip/hip_runtime.h>

// ---------------- problem constants ----------------
#define E_      8
#define D_      272       // input dim (K), padded to KP
#define H_      100       // hidden dim (N of GEMM), padded to HP
#define C_      2
#define N_      65536
#define KP      288       // 9 * 32
#define HP      112       // 7 * 16
#define KT      9         // KP / 32
#define NT      7         // HP / 16
#define ROWS_WG 64        // rows per workgroup (4 waves x 16)
#define THREADS 128
#define NEG_BIG_ (-1.0e30f)

typedef __attribute__((ext_vector_type(16))) __bf16         v16bf;
typedef __attribute__((ext_vector_type(8)))  float          v8f;
typedef __attribute__((ext_vector_type(8)))  unsigned short u16x8;
typedef __attribute__((ext_vector_type(16))) unsigned short u16x16;
typedef __attribute__((ext_vector_type(4)))  unsigned short u16x4;

// round-to-nearest-even f32 -> bf16 (bit pattern)
__device__ __forceinline__ unsigned short f2bf(float f) {
  unsigned u = __float_as_uint(f);
  u += 0x7FFFu + ((u >> 16) & 1u);
  return (unsigned short)(u >> 16);
}

// ---------------------------------------------------------------------------
// Prep: transpose+pad Ws/Wd into bf16 scratch: wt[s][e][n(HP)][k(KP)]
// Needs 2*E*HP*KP*2 = 1,032,192 bytes of d_ws.
// ---------------------------------------------------------------------------
__global__ void prep_weights_bf16(const float* __restrict__ Ws,
                                  const float* __restrict__ Wd,
                                  unsigned short* __restrict__ wt) {
  int idx = blockIdx.x * blockDim.x + threadIdx.x;
  const int total = 2 * E_ * HP * KP;
  if (idx >= total) return;
  int k = idx % KP;
  int n = (idx / KP) % HP;
  int e = (idx / (KP * HP)) % E_;
  int s = idx / (KP * HP * E_);
  float v = 0.0f;
  if (k < D_ && n < H_) {
    const float* W = s ? Wd : Ws;
    v = W[((size_t)e * D_ + k) * H_ + n];
  }
  wt[idx] = f2bf(v);
}

// ---------------------------------------------------------------------------
// One GEMM pass over N-tiles [NTB, NTB+NTC) for all K-tiles.
// A fragments (16-bit 16x32 layout): lane half g=lane>>4 holds K chunks
// [kt*32+8g .. +7] and [kt*32+8g+16 .. +23] for row M = lane&15.
// B fragments (16-bit 32x16 layout): lane half holds the same K chunks as A,
// [kt*32+8g .. +7] and [kt*32+8g+16 .. +23], for column N = lane&15  (weights stored N-major in LDS).
// ---------------------------------------------------------------------------
template <int NTB, int NTC>
__device__ __forceinline__ void gemm_pass(
    const unsigned short* shA0, const unsigned short* shA1,
    const unsigned short* shA2, const unsigned short* shWs,
    const unsigned short* shWd, int lane, int e,
    const float* __restrict__ bs, const float* __restrict__ bd,
    const float* __restrict__ Wo,
    float predp[8][2], float predn[8][2]) {
  v8f accS[NTC], accP[NTC], accN[NTC];
  const v8f zero = {0.f, 0.f, 0.f, 0.f, 0.f, 0.f, 0.f, 0.f};
#pragma unroll
  for (int j = 0; j < NTC; ++j) { accS[j] = zero; accP[j] = zero; accN[j] = zero; }

  const int rA   = lane & 15;
  const int kb8  = (lane >> 4) * 8;
  const int kb16 = (lane >> 4) * 16;

#pragma unroll
  for (int kt = 0; kt < KT; ++kt) {
    const int ka = kt * 32 + kb8;
    u16x8 a0lo = *(const u16x8*)(shA0 + rA * KP + ka);
    u16x8 a0hi = *(const u16x8*)(shA0 + rA * KP + ka + 16);
    u16x8 a1lo = *(const u16x8*)(shA1 + rA * KP + ka);
    u16x8 a1hi = *(const u16x8*)(shA1 + rA * KP + ka + 16);
    u16x8 a2lo = *(const u16x8*)(shA2 + rA * KP + ka);
    u16x8 a2hi = *(const u16x8*)(shA2 + rA * KP + ka + 16);
    u16x16 aSu = __builtin_shufflevector(a0lo, a0hi, 0,1,2,3,4,5,6,7,8,9,10,11,12,13,14,15);
    u16x16 aPu = __builtin_shufflevector(a1lo, a1hi, 0,1,2,3,4,5,6,7,8,9,10,11,12,13,14,15);
    u16x16 aNu = __builtin_shufflevector(a2lo, a2hi, 0,1,2,3,4,5,6,7,8,9,10,11,12,13,14,15);
    v16bf aS = __builtin_bit_cast(v16bf, aSu);
    v16bf aP = __builtin_bit_cast(v16bf, aPu);
    v16bf aN = __builtin_bit_cast(v16bf, aNu);

#pragma unroll
    for (int j = 0; j < NTC; ++j) {
      const int n  = (NTB + j) * 16 + rA;
      const int kb = kt * 32 + kb8;
      u16x8 bslo = *(const u16x8*)(shWs + n * KP + kb);
      u16x8 bshi = *(const u16x8*)(shWs + n * KP + kb + 16);
      u16x8 bdlo = *(const u16x8*)(shWd + n * KP + kb);
      u16x8 bdhi = *(const u16x8*)(shWd + n * KP + kb + 16);
      u16x16 bSu = __builtin_shufflevector(bslo, bshi, 0,1,2,3,4,5,6,7,8,9,10,11,12,13,14,15);
      u16x16 bDu = __builtin_shufflevector(bdlo, bdhi, 0,1,2,3,4,5,6,7,8,9,10,11,12,13,14,15);
      v16bf bS = __builtin_bit_cast(v16bf, bSu);
      v16bf bD = __builtin_bit_cast(v16bf, bDu);
      accS[j] = __builtin_amdgcn_wmma_f32_16x16x32_bf16(false, aS, false, bS, (short)0, accS[j], false, false);
      accP[j] = __builtin_amdgcn_wmma_f32_16x16x32_bf16(false, aP, false, bD, (short)0, accP[j], false, false);
      accN[j] = __builtin_amdgcn_wmma_f32_16x16x32_bf16(false, aN, false, bD, (short)0, accN[j], false, false);
    }
  }

  // epilogue: bias + relu + tiny Wo projection (C=2), in C/D register layout:
  // acc[v] on lane l is element (row = v + 8*(l>>4), col n = nt*16 + (l&15))
#pragma unroll
  for (int j = 0; j < NTC; ++j) {
    const int n = (NTB + j) * 16 + rA;
    float bsum = 0.f, wo0 = 0.f, wo1 = 0.f;
    if (n < H_) {
      bsum = bs[e * H_ + n] + bd[e * H_ + n];
      wo0  = Wo[((size_t)e * H_ + n) * C_ + 0];
      wo1  = Wo[((size_t)e * H_ + n) * C_ + 1];
    }
#pragma unroll
    for (int v = 0; v < 8; ++v) {
      float s  = accS[j][v];
      float hp = fmaxf(s + accP[j][v] + bsum, 0.f);
      float hn = fmaxf(s + accN[j][v] + bsum, 0.f);
      predp[v][0] += hp * wo0;  predp[v][1] += hp * wo1;
      predn[v][0] += hn * wo0;  predn[v][1] += hn * wo1;
    }
  }
}

// ---------------------------------------------------------------------------
// Main kernel: 128 threads = 4 waves, each wave owns a 16-row tile.
// LDS: sh_h[3][64][KP] bf16 (110592 B) + sh_w[2][HP][KP] bf16 (129024 B)
// ---------------------------------------------------------------------------
__global__ __launch_bounds__(THREADS, 1)
void hetero_edge_kernel(const float* __restrict__ h,
                        const float* __restrict__ bs,
                        const float* __restrict__ bd,
                        const float* __restrict__ Wo,
                        const float* __restrict__ bo,
                        const int* __restrict__ mask,
                        const unsigned short* __restrict__ wt,
                        float* __restrict__ out) {
  extern __shared__ char smem[];
  unsigned short* sh_h = (unsigned short*)smem;           // [3][ROWS_WG][KP]
  unsigned short* sh_w = sh_h + 3 * ROWS_WG * KP;         // [2][HP][KP]

  const int tid     = threadIdx.x;
  const int lane    = tid & 31;
  const int wr      = tid >> 5;                 // wave id: row tile 0..3
  const int rowbase = blockIdx.x * ROWS_WG;

  // ---- stage h_src / h_pos / h_neg as zero-padded bf16 rows in LDS ----
#pragma unroll
  for (int m = 0; m < 3; ++m) {
    const float* hsrc = h + (size_t)(m * N_ + rowbase) * D_;
    for (int idx = tid; idx < ROWS_WG * (KP / 4); idx += THREADS) {
      int r  = idx / (KP / 4);
      int k4 = (idx % (KP / 4)) * 4;
      float4 v = make_float4(0.f, 0.f, 0.f, 0.f);
      if (k4 < D_) v = *(const float4*)(hsrc + (size_t)r * D_ + k4);
      u16x4 b;
      b.x = f2bf(v.x); b.y = f2bf(v.y); b.z = f2bf(v.z); b.w = f2bf(v.w);
      *(u16x4*)(sh_h + (m * ROWS_WG + r) * KP + k4) = b;
    }
  }
  __syncthreads();

  float pmaxp[8][2], pmaxn[8][2];
  int   panyp[8], panyn[8];
#pragma unroll
  for (int v = 0; v < 8; ++v) {
    panyp[v] = 0; panyn[v] = 0;
    pmaxp[v][0] = pmaxp[v][1] = NEG_BIG_;
    pmaxn[v][0] = pmaxn[v][1] = NEG_BIG_;
  }

  const unsigned short* shA0 = sh_h + (0 * ROWS_WG + wr * 16) * KP;
  const unsigned short* shA1 = sh_h + (1 * ROWS_WG + wr * 16) * KP;
  const unsigned short* shA2 = sh_h + (2 * ROWS_WG + wr * 16) * KP;
  const unsigned short* shWs = sh_w;
  const unsigned short* shWd = sh_w + HP * KP;

  for (int e = 0; e < E_; ++e) {
    // stage this edge type's transposed bf16 weights (L2-hot, wide copies)
    {
      const int QW = HP * KP / 8;               // uint4 (8 bf16) per matrix
      for (int idx = tid; idx < 2 * QW; idx += THREADS) {
        int s = idx / QW, j = idx % QW;
        ((uint4*)sh_w)[(size_t)s * QW + j] =
            ((const uint4*)(wt + (size_t)(s * E_ + e) * HP * KP))[j];
      }
      if (e + 1 < E_) {                         // global_prefetch_b8 next e
        __builtin_prefetch(wt + (size_t)(e + 1) * HP * KP + tid * 64, 0, 1);
        __builtin_prefetch(wt + (size_t)(E_ + e + 1) * HP * KP + tid * 64, 0, 1);
      }
    }
    __syncthreads();

    float predp[8][2] = {{0.f}}, predn[8][2] = {{0.f}};
    gemm_pass<0, 4>(shA0, shA1, shA2, shWs, shWd, lane, e, bs, bd, Wo, predp, predn);
    gemm_pass<4, 3>(shA0, shA1, shA2, shWs, shWd, lane, e, bs, bd, Wo, predp, predn);

    // reduce partial dot products across the 16 lanes of each half-wave
#pragma unroll
    for (int v = 0; v < 8; ++v) {
#pragma unroll
      for (int c = 0; c < 2; ++c) {
        float sp = predp[v][c], sn = predn[v][c];
        for (int m = 1; m < 16; m <<= 1) {
          sp += __shfl_xor(sp, m, 32);
          sn += __shfl_xor(sn, m, 32);
        }
        predp[v][c] = sp; predn[v][c] = sn;
      }
    }

    // masked running max over edge types
    const int rgh = rowbase + wr * 16 + (lane >> 4) * 8;
#pragma unroll
    for (int v = 0; v < 8; ++v) {
      int rp = rgh + v;
      int mp = mask[(size_t)rp * E_ + e];
      int mn = mask[((size_t)N_ + rp) * E_ + e];
#pragma unroll
      for (int c = 0; c < 2; ++c) {
        float cp = predp[v][c] + bo[e * C_ + c];
        float cn = predn[v][c] + bo[e * C_ + c];
        pmaxp[v][c] = mp ? fmaxf(pmaxp[v][c], cp) : pmaxp[v][c];
        pmaxn[v][c] = mn ? fmaxf(pmaxn[v][c], cn) : pmaxn[v][c];
      }
      panyp[v] |= mp; panyn[v] |= mn;
    }
    __syncthreads();   // protect sh_w before next edge type's staging
  }

  // write: pos_out (N,2) then neg_out (N,2); lanes 0 and 16 own the halves
  if ((lane & 15) == 0) {
    const int rgh = rowbase + wr * 16 + (lane >> 4) * 8;
#pragma unroll
    for (int v = 0; v < 8; ++v) {
      int rp = rgh + v;
#pragma unroll
      for (int c = 0; c < 2; ++c) {
        out[(size_t)rp * C_ + c]                 = panyp[v] ? pmaxp[v][c] : 0.0f;
        out[(size_t)N_ * C_ + (size_t)rp * C_ + c] = panyn[v] ? pmaxn[v][c] : 0.0f;
      }
    }
  }
}

// ---------------------------------------------------------------------------
extern "C" void kernel_launch(void* const* d_in, const int* in_sizes, int n_in,
                              void* d_out, int out_size, void* d_ws, size_t ws_size,
                              hipStream_t stream) {
  const float* h    = (const float*)d_in[0];
  const float* Ws   = (const float*)d_in[1];
  const float* bs   = (const float*)d_in[2];
  const float* Wd   = (const float*)d_in[3];
  const float* bd   = (const float*)d_in[4];
  const float* Wo   = (const float*)d_in[5];
  const float* bo   = (const float*)d_in[6];
  const int*   mask = (const int*)d_in[7];
  float* out = (float*)d_out;
  unsigned short* wt = (unsigned short*)d_ws;   // 2*E*HP*KP*2 ~= 1.0 MB scratch

  const int total = 2 * E_ * HP * KP;
  prep_weights_bf16<<<(total + 255) / 256, 256, 0, stream>>>(Ws, Wd, wt);

  const size_t smem = (size_t)(3 * ROWS_WG * KP + 2 * HP * KP) * sizeof(unsigned short);
  (void)hipFuncSetAttribute((const void*)hetero_edge_kernel,
                            hipFuncAttributeMaxDynamicSharedMemorySize, (int)smem);
  hetero_edge_kernel<<<N_ / ROWS_WG, THREADS, smem, stream>>>(
      h, bs, bd, Wo, bo, mask, wt, out);
}
